// InfiniteBrain_8873402433626
// MI455X (gfx1250) — matrix operand order repaired and checked by reading, not compiled, not run on hardware
//
#include <hip/hip_runtime.h>
#include <hip/hip_bf16.h>

typedef __bf16 bf16;
typedef bf16  v16bf __attribute__((ext_vector_type(16)));
typedef bf16  v8bf  __attribute__((ext_vector_type(8)));
typedef float v8f   __attribute__((ext_vector_type(8)));

#define TT     1024
#define NROW   8192      // B*T
#define DD     256
#define CC     32768
#define VOCABN 8192
#define DEMB   64

// Build a 16-element bf16 WMMA fragment from two contiguous 8-element (16B) runs.
__device__ __forceinline__ v16bf frag2(const bf16* p0, const bf16* p1) {
  v8bf lo = *(const v8bf*)p0;
  v8bf hi = *(const v8bf*)p1;
  v16bf r;
#pragma unroll
  for (int i = 0; i < 8; ++i) { r[i] = lo[i]; r[i + 8] = hi[i]; }
  return r;
}

__global__ void ib_conv_cells(const float* __restrict__ src, bf16* __restrict__ dst, int n) {
  int i = blockIdx.x * blockDim.x + threadIdx.x;
  if (i < n) dst[i] = (bf16)src[i];
}

// W_out (D,VOCAB) -> bf16 transposed (VOCAB,D): B-fragments load contiguously.
__global__ void ib_conv_woutT(const float* __restrict__ W, bf16* __restrict__ WT) {
  int i = blockIdx.x * blockDim.x + threadIdx.x;   // i = k*VOCAB + n
  int k = i >> 13;
  int n = i & (VOCABN - 1);
  WT[n * DD + k] = (bf16)W[i];
}

// W_gate (D,D) -> bf16 transposed (D,D).
__global__ void ib_conv_wgT(const float* __restrict__ W, bf16* __restrict__ WT) {
  int i = blockIdx.x * blockDim.x + threadIdx.x;   // i = k*D + n
  int k = i >> 8;
  int n = i & (DD - 1);
  WT[n * DD + k] = (bf16)W[i];
}

__global__ void ib_embed_proj(const int* __restrict__ x, const float* __restrict__ emb,
                              const float* __restrict__ Wp, const float* __restrict__ bp,
                              float* __restrict__ P, bf16* __restrict__ Pbf) {
  __shared__ float es[DEMB];
  const int row = blockIdx.x;
  const int d   = threadIdx.x;
  if (d < DEMB) es[d] = emb[x[row] * DEMB + d];
  __syncthreads();
  float acc = bp[d];
#pragma unroll 8
  for (int e = 0; e < DEMB; ++e) acc += es[e] * Wp[e * DD + d];
  P[row * DD + d]   = acc;
  Pbf[row * DD + d] = (bf16)acc;
}

// G = sigmoid(P @ W_gate + b_gate), WMMA, 32 rows per WG.
__global__ void __launch_bounds__(256) ib_gate_wmma(
    const bf16* __restrict__ Pbf, const bf16* __restrict__ WgT,
    const float* __restrict__ bg, float* __restrict__ G) {
  const int tid  = threadIdx.x;
  const int wave = tid >> 5, lane = tid & 31;
  const int half = lane >> 4, lm = lane & 15;
  const int rowbase = blockIdx.x * 32;

  v16bf aF0[8], aF1[8];
  {
    const bf16* a0 = Pbf + (rowbase + lm) * DD + half * 8;
    const bf16* a1 = a0 + 16 * DD;
#pragma unroll
    for (int kc = 0; kc < 8; ++kc) {
      aF0[kc] = frag2(a0 + kc * 32, a0 + kc * 32 + 16);
      aF1[kc] = frag2(a1 + kc * 32, a1 + kc * 32 + 16);
    }
  }
  for (int it = 0; it < 2; ++it) {            // 16 col tiles over 8 waves
    const int colbase = (wave * 2 + it) * 16;
    const bf16* bbase = WgT + (colbase + lm) * DD + half * 8;
    v8f c0 = {0.f,0.f,0.f,0.f,0.f,0.f,0.f,0.f};
    v8f c1 = c0;
#pragma unroll
    for (int kc = 0; kc < 8; ++kc) {
      v16bf bF = frag2(bbase + kc * 32, bbase + kc * 32 + 16);
      c0 = __builtin_amdgcn_wmma_f32_16x16x32_bf16(false, aF0[kc], false, bF, (short)0, c0, false, false);
      c1 = __builtin_amdgcn_wmma_f32_16x16x32_bf16(false, aF1[kc], false, bF, (short)0, c1, false, false);
    }
    const float b = bg[colbase + lm];
    float* g0 = G + (rowbase + half * 8) * DD + colbase + lm;
#pragma unroll
    for (int v = 0; v < 8; ++v) {
      g0[(v)      * DD] = 1.0f / (1.0f + __expf(-(c0[v] + b)));
      g0[(v + 16) * DD] = 1.0f / (1.0f + __expf(-(c1[v] + b)));
    }
  }
}

// Fused: SIM = P @ cells^T (+cell_bias), streaming exact top-8 per row,
// softmax(min(v,50)/2), upd = sum_k w_k * cells[idx_k].  32 rows per WG.
__global__ void __launch_bounds__(256) ib_sim_topk_upd(
    const bf16* __restrict__ Pbf, const bf16* __restrict__ Cbf,
    const float* __restrict__ cells, const float* __restrict__ cbias,
    float* __restrict__ U) {
  __shared__ float simbuf[8][32 * 17];        // stride 17: conflict-free readback
  __shared__ float mv[256 * 8];
  __shared__ int   mi[256 * 8];
  __shared__ float wrow[32][8];
  __shared__ int   irow[32][8];

  const int tid  = threadIdx.x;
  const int wave = tid >> 5, lane = tid & 31;
  const int half = lane >> 4, lm = lane & 15;
  const int rowbase = blockIdx.x * 32;

  v16bf aF0[8], aF1[8];
  {
    const bf16* a0 = Pbf + (rowbase + lm) * DD + half * 8;
    const bf16* a1 = a0 + 16 * DD;
#pragma unroll
    for (int kc = 0; kc < 8; ++kc) {
      aF0[kc] = frag2(a0 + kc * 32, a0 + kc * 32 + 16);
      aF1[kc] = frag2(a1 + kc * 32, a1 + kc * 32 + 16);
    }
  }

  float tv[8]; int ti[8];
#pragma unroll
  for (int k = 0; k < 8; ++k) { tv[k] = -3.0e38f; ti[k] = 0; }

  float* sb = simbuf[wave];

  for (int it = 0; it < 256; ++it) {
    const int tile    = wave + it * 8;
    const int colbase = tile * 16;
    const bf16* bbase = Cbf + (colbase + lm) * DD + half * 8;
    if (it < 255)
      __builtin_prefetch(Cbf + (colbase + 128 + lm) * DD + half * 16, 0, 3);

    v8f c0 = {0.f,0.f,0.f,0.f,0.f,0.f,0.f,0.f};
    v8f c1 = c0;
#pragma unroll
    for (int kc = 0; kc < 8; ++kc) {
      v16bf bF = frag2(bbase + kc * 32, bbase + kc * 32 + 16);
      c0 = __builtin_amdgcn_wmma_f32_16x16x32_bf16(false, aF0[kc], false, bF, (short)0, c0, false, false);
      c1 = __builtin_amdgcn_wmma_f32_16x16x32_bf16(false, aF1[kc], false, bF, (short)0, c1, false, false);
    }
    const float bias = cbias[colbase + lm];
#pragma unroll
    for (int v = 0; v < 8; ++v) {             // D layout: row = v + half*8, col = lm
      sb[(v + half * 8)      * 17 + lm] = c0[v] + bias;
      sb[(v + half * 8 + 16) * 17 + lm] = c1[v] + bias;
    }
    __builtin_amdgcn_wave_barrier();          // per-wave buffer; LDS is in-order per wave

    // one lane per row scans 16 columns, keeps sorted-ascending top-8
#pragma unroll
    for (int j = 0; j < 16; ++j) {
      const float val = sb[lane * 17 + j];
      const int gidx  = colbase + j;
      if (val > tv[0]) {
        int p = 0;
#pragma unroll
        for (int q = 0; q < 7; ++q)
          if (val > tv[q + 1]) { tv[q] = tv[q + 1]; ti[q] = ti[q + 1]; p = q + 1; }
        tv[p] = val; ti[p] = gidx;
      }
    }
    __builtin_amdgcn_wave_barrier();
  }

  // Dump per-lane lists; exact merge of 64 candidates per row.
#pragma unroll
  for (int k = 0; k < 8; ++k) { mv[tid * 8 + k] = tv[k]; mi[tid * 8 + k] = ti[k]; }
  __syncthreads();

  if (tid < 32) {
    float fv[8]; int fi[8];
#pragma unroll
    for (int k = 0; k < 8; ++k) { fv[k] = -3.0e38f; fi[k] = 0; }
    for (int w = 0; w < 8; ++w) {
      const int src = w * 32 + tid;           // lane `tid` of wave w owned row `tid`
      for (int k = 0; k < 8; ++k) {
        const float v = mv[src * 8 + k];
        const int  gi = mi[src * 8 + k];
        if (v > fv[0]) {
          int p = 0;
          for (int q = 0; q < 7; ++q)
            if (v > fv[q + 1]) { fv[q] = fv[q + 1]; fi[q] = fi[q + 1]; p = q + 1; }
          fv[p] = v; fi[p] = gi;
        }
      }
    }
    const float m = fminf(fv[7], 50.0f) * 0.5f;   // list sorted ascending
    float s = 0.0f, wk[8];
#pragma unroll
    for (int k = 0; k < 8; ++k) { wk[k] = __expf(fminf(fv[k], 50.0f) * 0.5f - m); s += wk[k]; }
    const float inv = 1.0f / s;
#pragma unroll
    for (int k = 0; k < 8; ++k) { wrow[tid][k] = wk[k] * inv; irow[tid][k] = fi[k]; }
  }
  __syncthreads();

  // upd: 32 rows x 256 dims over 256 threads (f32 cells for accuracy).
  {
    const int row   = tid >> 3;
    const int dbase = (tid & 7) * 32;
    float w[8]; int id[8];
#pragma unroll
    for (int k = 0; k < 8; ++k) { w[k] = wrow[row][k]; id[k] = irow[row][k]; }
    for (int d = dbase; d < dbase + 32; ++d) {
      float acc = 0.0f;
#pragma unroll
      for (int k = 0; k < 8; ++k) acc += w[k] * cells[id[k] * DD + d];
      U[(rowbase + row) * DD + d] = acc;
    }
  }
}

// Truly sequential part: h_t = (1-g)h + g*u, clip(+-10). 2048 independent lanes.
__global__ void ib_scan(const float* __restrict__ G, const float* __restrict__ U,
                        float* __restrict__ H) {
  const int t = blockIdx.x * blockDim.x + threadIdx.x;   // 0..2047
  const int b = t >> 8, d = t & 255;
  const float* g = G + b * (TT * DD) + d;
  const float* u = U + b * (TT * DD) + d;
  float* h = H + b * (TT * DD) + d;
  float hv = 0.0f;
#pragma unroll 4
  for (int s = 0; s < TT; ++s) {
    const float gv = g[s * DD], uv = u[s * DD];
    hv = (1.0f - gv) * hv + gv * uv;
    hv = fminf(fmaxf(hv, -10.0f), 10.0f);
    h[s * DD] = hv;
  }
}

// LayerNorm of z = P + H, wave per row, coalesced, shfl reductions.
__global__ void ib_ln(const float* __restrict__ P, const float* __restrict__ H,
                      const float* __restrict__ gamma, const float* __restrict__ beta,
                      bf16* __restrict__ Znbf) {
  const int row  = blockIdx.x * 8 + (threadIdx.x >> 5);
  const int lane = threadIdx.x & 31;
  float zv[8];
  float s = 0.f, s2 = 0.f;
#pragma unroll
  for (int i = 0; i < 8; ++i) {
    const int d = lane + i * 32;
    const float z = P[row * DD + d] + H[row * DD + d];
    zv[i] = z; s += z; s2 += z * z;
  }
#pragma unroll
  for (int off = 16; off >= 1; off >>= 1) {
    s  += __shfl_xor(s,  off, 32);
    s2 += __shfl_xor(s2, off, 32);
  }
  const float mu  = s * (1.0f / DD);
  const float var = s2 * (1.0f / DD) - mu * mu;
  const float rs  = rsqrtf(var + 1e-5f);
#pragma unroll
  for (int i = 0; i < 8; ++i) {
    const int d = lane + i * 32;
    Znbf[row * DD + d] = (bf16)((zv[i] - mu) * rs * gamma[d] + beta[d]);
  }
}

// out = Zn @ W_out + b_out : 32 rows per WG, A loaded once, 64 column tiles/wave.
__global__ void __launch_bounds__(256) ib_out_gemm(
    const bf16* __restrict__ Znbf, const bf16* __restrict__ WT,
    const float* __restrict__ bo, float* __restrict__ out) {
  const int tid  = threadIdx.x;
  const int wave = tid >> 5, lane = tid & 31;
  const int half = lane >> 4, lm = lane & 15;
  const int rowbase = blockIdx.x * 32;

  v16bf aF0[8], aF1[8];
  {
    const bf16* a0 = Znbf + (rowbase + lm) * DD + half * 8;
    const bf16* a1 = a0 + 16 * DD;
#pragma unroll
    for (int kc = 0; kc < 8; ++kc) {
      aF0[kc] = frag2(a0 + kc * 32, a0 + kc * 32 + 16);
      aF1[kc] = frag2(a1 + kc * 32, a1 + kc * 32 + 16);
    }
  }
  for (int it = 0; it < 64; ++it) {
    const int tile = wave + it * 8;
    const int colbase = tile * 16;
    const bf16* bbase = WT + (colbase + lm) * DD + half * 8;
    v8f c0 = {0.f,0.f,0.f,0.f,0.f,0.f,0.f,0.f};
    v8f c1 = c0;
#pragma unroll
    for (int kc = 0; kc < 8; ++kc) {
      v16bf bF = frag2(bbase + kc * 32, bbase + kc * 32 + 16);
      c0 = __builtin_amdgcn_wmma_f32_16x16x32_bf16(false, aF0[kc], false, bF, (short)0, c0, false, false);
      c1 = __builtin_amdgcn_wmma_f32_16x16x32_bf16(false, aF1[kc], false, bF, (short)0, c1, false, false);
    }
    const float bias = bo[colbase + lm];
    float* obase = out + (size_t)(rowbase + half * 8) * VOCABN + colbase + lm;
#pragma unroll
    for (int v = 0; v < 8; ++v) {
      obase[(size_t)v        * VOCABN] = c0[v] + bias;
      obase[(size_t)(v + 16) * VOCABN] = c1[v] + bias;
    }
  }
}

extern "C" void kernel_launch(void* const* d_in, const int* in_sizes, int n_in,
                              void* d_out, int out_size, void* d_ws, size_t ws_size,
                              hipStream_t stream) {
  (void)in_sizes; (void)n_in; (void)out_size; (void)ws_size;
  const int*   x     = (const int*)d_in[0];
  const float* emb   = (const float*)d_in[1];
  const float* Wp    = (const float*)d_in[2];
  const float* bp    = (const float*)d_in[3];
  const float* cells = (const float*)d_in[4];
  const float* cbias = (const float*)d_in[5];
  const float* Wg    = (const float*)d_in[6];
  const float* bg    = (const float*)d_in[7];
  const float* gamma = (const float*)d_in[8];
  const float* beta  = (const float*)d_in[9];
  const float* Wo    = (const float*)d_in[10];
  const float* bo    = (const float*)d_in[11];
  float* out = (float*)d_out;

  char* ws = (char*)d_ws;                       // ~60 MB fixed layout
  float* P    = (float*)(ws + 0);               // 8 MB
  float* G    = (float*)(ws + 8388608);         // 8 MB
  float* U    = (float*)(ws + 16777216);        // 8 MB
  float* H    = (float*)(ws + 25165824);        // 8 MB
  bf16*  Pbf  = (bf16*) (ws + 33554432);        // 4 MB
  bf16*  Znbf = (bf16*) (ws + 37748736);        // 4 MB
  bf16*  Cbf  = (bf16*) (ws + 41943040);        // 16 MB
  bf16*  WT   = (bf16*) (ws + 58720256);        // 4 MB
  bf16*  WgT  = (bf16*) (ws + 62914560);        // 128 KB

  ib_conv_cells<<<(CC * DD) / 256, 256, 0, stream>>>(cells, Cbf, CC * DD);
  ib_conv_woutT<<<(DD * VOCABN) / 256, 256, 0, stream>>>(Wo, WT);
  ib_conv_wgT<<<(DD * DD) / 256, 256, 0, stream>>>(Wg, WgT);
  ib_embed_proj<<<NROW, 256, 0, stream>>>(x, emb, Wp, bp, P, Pbf);
  ib_gate_wmma<<<NROW / 32, 256, 0, stream>>>(Pbf, WgT, bg, G);
  ib_sim_topk_upd<<<NROW / 32, 256, 0, stream>>>(Pbf, Cbf, cells, cbias, U);
  ib_scan<<<8, 256, 0, stream>>>(G, U, H);
  ib_ln<<<NROW / 256, 256, 0, stream>>>(P, H, gamma, beta, Znbf);
  ib_out_gemm<<<NROW / 32, 256, 0, stream>>>(Znbf, WT, bo, out);
}
